// SmolMoEDecoderLayer_72361609003122
// MI455X (gfx1250) — compile-verified, matrix operand order checked by reading, not run on hardware
//
#include <hip/hip_runtime.h>
#include <hip/hip_bf16.h>
#include <math.h>

// ---------------- problem constants ----------------
constexpr int CB = 2, CS = 1024, CH = 2048, CNH = 32, CHD = 64, CE = 8, CM = 1024;
constexpr int CT = CB * CS; // 2048 tokens

typedef __attribute__((ext_vector_type(16))) _Float16 v16h;
typedef __attribute__((ext_vector_type(8)))  _Float16 v8h;
typedef __attribute__((ext_vector_type(8)))  float    v8f;
typedef __attribute__((ext_vector_type(4)))  int      v4i;

#ifndef __has_builtin
#define __has_builtin(x) 0
#endif
#if __has_builtin(__builtin_amdgcn_global_load_async_to_lds_b128) && \
    __has_builtin(__builtin_amdgcn_s_wait_asynccnt)
#define ASYNC_LDS 1
#else
#define ASYNC_LDS 0
#endif

__device__ __forceinline__ v16h make_frag(const _Float16* lo, const _Float16* hi) {
    v8h a = *(const v8h*)lo;
    v8h b = *(const v8h*)hi;
    v16h r;
#pragma unroll
    for (int i = 0; i < 8; ++i) { r[i] = a[i]; r[i + 8] = b[i]; }
    return r;
}

__device__ __forceinline__ v8f wmma16(v16h a, v16h b, v8f c) {
    return __builtin_amdgcn_wmma_f32_16x16x32_f16(false, a, false, b, (short)0, c, false, false);
}

// 16-byte global -> LDS copy; async DMA on CDNA5 (ASYNCcnt-tracked), sync fallback otherwise.
__device__ __forceinline__ void copy16_async(_Float16* lds, const _Float16* g) {
#if ASYNC_LDS
    __builtin_amdgcn_global_load_async_to_lds_b128(
        (__attribute__((address_space(1))) v4i*)(uintptr_t)g,
        (__attribute__((address_space(3))) v4i*)(unsigned)(uintptr_t)lds,
        0, 0);
#else
    *(v8h*)lds = *(const v8h*)g;
#endif
}

#if ASYNC_LDS
#define WAIT_ASYNC(n) do { __builtin_amdgcn_s_wait_asynccnt(n); asm volatile("" ::: "memory"); } while (0)
#else
#define WAIT_ASYNC(n) do { asm volatile("" ::: "memory"); } while (0)
#endif

// ---------------- RMSNorm ----------------
__global__ __launch_bounds__(256) void rmsnorm_kernel(const float* __restrict__ x,
                                                      const float* __restrict__ w,
                                                      float* __restrict__ out, int Hdim) {
    __shared__ float red[256];
    int t = blockIdx.x, tid = threadIdx.x;
    const float* row = x + (size_t)t * Hdim;
    float s = 0.f;
    for (int i = tid; i < Hdim; i += 256) { float v = row[i]; s += v * v; }
    red[tid] = s; __syncthreads();
    for (int off = 128; off > 0; off >>= 1) {
        if (tid < off) red[tid] += red[tid + off];
        __syncthreads();
    }
    float inv = rsqrtf(red[0] / (float)Hdim + 1e-5f);
    for (int i = tid; i < Hdim; i += 256)
        out[(size_t)t * Hdim + i] = row[i] * inv * w[i];
}

// ---------------- f16-WMMA GEMM: C[M,N] = A[M,K] x B[K,N] ----------------
// 128x128 block tile, 256 threads (8 waves), each wave 32x64 (2x4 WMMA tiles).
// f32 inputs converted to f16 while staging to LDS; f32 accumulate.
// EPI: 0 = plain store, 1 = += addend, 2 = C += rowscale[m]*AB (accumulate).
template <int EPI>
__global__ __launch_bounds__(256) void gemm_kernel(const float* __restrict__ A,
                                                   const float* __restrict__ B,
                                                   float* __restrict__ C,
                                                   int Mdim, int Ndim, int Kdim,
                                                   const float* __restrict__ addend,
                                                   const float* __restrict__ rowscale,
                                                   int rsStride) {
    __shared__ __align__(16) _Float16 lA[128][40];   // [m][k], row stride 80B (16B multiple)
    __shared__ __align__(16) _Float16 lB[128][40];   // transposed: [n][k]
    int tid = threadIdx.x;
    int wv = tid >> 5, lane = tid & 31;
    int m0 = blockIdx.y * 128, n0 = blockIdx.x * 128;
    int r0 = (wv >> 1) * 32, c0 = (wv & 1) * 64;
    int lm = lane & 15, koff = (lane >> 4) * 8;

    v8f acc[2][4] = {};

    for (int k0 = 0; k0 < Kdim; k0 += 32) {
        if (k0 + 32 < Kdim) {   // prefetch next K-slab (global_prefetch_b8)
            __builtin_prefetch(A + (size_t)(m0 + (tid >> 1)) * Kdim + k0 + 32, 0, 3);
            __builtin_prefetch(B + (size_t)(k0 + 32 + (tid >> 3)) * Ndim + n0, 0, 3);
        }
#pragma unroll
        for (int i = 0; i < 4; ++i) {
            int c = tid + 256 * i;              // 1024 float4 chunks per operand
            {   // A tile: 128 rows x 32 cols
                int r = c >> 3, cc = (c & 7) * 4;
                float4 f = *(const float4*)(A + (size_t)(m0 + r) * Kdim + k0 + cc);
                lA[r][cc + 0] = (_Float16)f.x; lA[r][cc + 1] = (_Float16)f.y;
                lA[r][cc + 2] = (_Float16)f.z; lA[r][cc + 3] = (_Float16)f.w;
            }
            {   // B tile: 32 rows x 128 cols, stored transposed [n][k]
                int r = c >> 5, cc = (c & 31) * 4;
                float4 f = *(const float4*)(B + (size_t)(k0 + r) * Ndim + n0 + cc);
                lB[cc + 0][r] = (_Float16)f.x; lB[cc + 1][r] = (_Float16)f.y;
                lB[cc + 2][r] = (_Float16)f.z; lB[cc + 3][r] = (_Float16)f.w;
            }
        }
        __syncthreads();

        v16h a[2], bf[4];
#pragma unroll
        for (int i = 0; i < 2; ++i)
            a[i] = make_frag(&lA[r0 + 16 * i + lm][koff], &lA[r0 + 16 * i + lm][16 + koff]);
#pragma unroll
        for (int j = 0; j < 4; ++j)
            bf[j] = make_frag(&lB[c0 + 16 * j + lm][koff], &lB[c0 + 16 * j + lm][16 + koff]);
#pragma unroll
        for (int i = 0; i < 2; ++i)
#pragma unroll
            for (int j = 0; j < 4; ++j)
                acc[i][j] = wmma16(a[i], bf[j], acc[i][j]);
        __syncthreads();
    }

#pragma unroll
    for (int i = 0; i < 2; ++i) {
#pragma unroll
        for (int v = 0; v < 8; ++v) {
            int gm = m0 + r0 + i * 16 + v + ((lane >> 4) << 3);
            float rs = (EPI == 2) ? rowscale[(size_t)gm * rsStride] : 0.f;
#pragma unroll
            for (int j = 0; j < 4; ++j) {
                int gn = n0 + c0 + j * 16 + lm;
                size_t idx = (size_t)gm * Ndim + gn;
                float val = acc[i][j][v];
                if (EPI == 1) val += addend[idx];
                if (EPI == 2) val = C[idx] + rs * val;
                C[idx] = val;
            }
        }
    }
}

// ---------------- RoPE + head-layout (+ V transpose), f32 -> f16 ----------------
__global__ __launch_bounds__(64) void rope_kernel(const float* __restrict__ q,
                                                  const float* __restrict__ k,
                                                  const float* __restrict__ v,
                                                  _Float16* __restrict__ qr,
                                                  _Float16* __restrict__ kr,
                                                  _Float16* __restrict__ vt) {
    int s = blockIdx.x, h = blockIdx.y, b = blockIdx.z, d = threadIdx.x;
    size_t src = (size_t)(b * CS + s) * CH + h * CHD + d;
    float invf = __expf(-(float)(d & 31) * (logf(10000.f) / 32.f));
    float ang = (float)s * invf;
    float sn, cs;
    __sincosf(ang, &sn, &cs);
    float qv = q[src], kv = k[src];
    float qo = (d < 32) ? -q[src + 32] : q[src - 32];
    float ko = (d < 32) ? -k[src + 32] : k[src - 32];
    size_t dst = ((size_t)(b * CNH + h) * CS + s) * CHD + d;
    qr[dst] = (_Float16)(qv * cs + qo * sn);
    kr[dst] = (_Float16)(kv * cs + ko * sn);
    vt[((size_t)(b * CNH + h) * CHD + d) * CS + s] = (_Float16)v[src];
}

// ---------------- causal attention, one wave per (b, head, 16-query tile) ----------------
// K and V tiles double-buffered into LDS with async global->LDS DMA (ASYNCcnt).
constexpr int SSROW = 1028;                        // f32 score row stride (padded)
constexpr int SPROW = 1032;                        // f16 prob row stride (16B-multiple)
constexpr size_t SS_BYTES = 16 * SSROW * sizeof(float);
constexpr size_t SP_BYTES = 16 * SPROW * sizeof(_Float16);
constexpr size_t SK_BYTES = 2 * 16 * 64 * sizeof(_Float16);   // 2 x (16 keys x 64 d)
constexpr size_t SV_BYTES = 2 * 64 * 32 * sizeof(_Float16);   // 2 x (64 d x 32 keys)
constexpr size_t ATTN_LDS = SS_BYTES + SP_BYTES + SK_BYTES + SV_BYTES;

__global__ __launch_bounds__(32) void attn_kernel(const _Float16* __restrict__ Qr,
                                                  const _Float16* __restrict__ Kr,
                                                  const _Float16* __restrict__ Vt,
                                                  float* __restrict__ ctx) {
    extern __shared__ char smem[];
    float*    sS = (float*)smem;                                   // [16][SSROW]
    _Float16* sP = (_Float16*)(smem + SS_BYTES);                   // [16][SPROW]
    _Float16* sK = (_Float16*)(smem + SS_BYTES + SP_BYTES);        // [2][16][64]
    _Float16* sV = (_Float16*)(smem + SS_BYTES + SP_BYTES + SK_BYTES); // [2][64][32]

    int qt = blockIdx.x, h = blockIdx.y, b = blockIdx.z;
    int q0 = qt * 16;
    int lane = threadIdx.x, lm = lane & 15, hf = lane >> 4, koff = hf * 8;

    const _Float16* Qh = Qr + (size_t)(b * CNH + h) * CS * CHD;   // [s][d]
    const _Float16* Kh = Kr + (size_t)(b * CNH + h) * CS * CHD;   // [s][d]
    const _Float16* Vh = Vt + (size_t)(b * CNH + h) * CHD * CS;   // [d][s]

    v16h qf[2];
#pragma unroll
    for (int ks = 0; ks < 2; ++ks) {
        const _Float16* p = Qh + (size_t)(q0 + lm) * CHD + ks * 32;
        qf[ks] = make_frag(p + koff, p + 16 + koff);
    }

    // --- stage helpers: 16B chunks, 4 async per lane for K tile, 8 for V tile ---
    auto stageK = [&](int kt) {                    // tile = 16 rows x 64 halfs (2KB)
        int buf = kt & 1;
#pragma unroll
        for (int i = 0; i < 4; ++i) {
            int c = lane + 32 * i;                 // 128 chunks
            int row = c >> 3, cc = (c & 7) * 8;
            copy16_async(sK + buf * 1024 + row * 64 + cc,
                         Kh + (size_t)(kt * 16 + row) * CHD + cc);
        }
    };
    auto stageV = [&](int ch) {                    // tile = 64 d-rows x 32 halfs (4KB)
        int buf = ch & 1;
#pragma unroll
        for (int i = 0; i < 8; ++i) {
            int c = lane + 32 * i;                 // 256 chunks
            int row = c >> 2, cc = (c & 3) * 8;
            copy16_async(sV + buf * 2048 + row * 32 + cc,
                         Vh + (size_t)row * CS + ch * 32 + cc);
        }
    };

    // scores: S = (Q K^T)/8, causal tiles only, K tile double-buffered
    stageK(0);
    for (int kt = 0; kt <= qt; ++kt) {
        int buf = kt & 1;
        if (kt + 1 <= qt) { stageK(kt + 1); WAIT_ASYNC(4); }
        else             { WAIT_ASYNC(0); }
        int j0 = kt * 16;
        v8f c = {};
#pragma unroll
        for (int ks = 0; ks < 2; ++ks) {
            v16h bf = make_frag(sK + buf * 1024 + lm * 64 + ks * 32 + koff,
                                sK + buf * 1024 + lm * 64 + ks * 32 + 16 + koff);
            c = wmma16(qf[ks], bf, c);
        }
#pragma unroll
        for (int v = 0; v < 8; ++v) {
            int row = v + hf * 8;
            int key = j0 + lm;
            float sc = c[v] * 0.125f;
            if (key > q0 + row) sc = -1e30f;
            sS[row * SSROW + key] = sc;
        }
    }
    __syncthreads();

    int kmax = (qt + 1) * 16;
    int nch  = (kmax + 31) >> 5;
    int kpad = nch * 32;

    for (int row = 0; row < 16; ++row) {
        int L = q0 + row + 1;
        float mx = -1e30f;
        for (int k = lane; k < L; k += 32) mx = fmaxf(mx, sS[row * SSROW + k]);
#pragma unroll
        for (int off = 16; off >= 1; off >>= 1) mx = fmaxf(mx, __shfl_xor(mx, off, 32));
        float sum = 0.f;
        for (int k = lane; k < L; k += 32) {
            float e = __expf(sS[row * SSROW + k] - mx);
            sum += e;
            sS[row * SSROW + k] = e;
        }
#pragma unroll
        for (int off = 16; off >= 1; off >>= 1) sum += __shfl_xor(sum, off, 32);
        float inv = 1.f / sum;
        for (int k = lane; k < kpad; k += 32)
            sP[row * SPROW + k] = (k < L) ? (_Float16)(sS[row * SSROW + k] * inv) : (_Float16)0.f;
    }
    __syncthreads();

    // O = P V (16 x 64), key chunks of 32, V tile double-buffered
    v8f oacc[4] = {};
    stageV(0);
    for (int ch = 0; ch < nch; ++ch) {
        int buf = ch & 1;
        if (ch + 1 < nch) { stageV(ch + 1); WAIT_ASYNC(8); }
        else              { WAIT_ASYNC(0); }
        int kk = ch * 32;
        v16h pf = make_frag(&sP[lm * SPROW + kk + koff], &sP[lm * SPROW + kk + 16 + koff]);
#pragma unroll
        for (int nt = 0; nt < 4; ++nt) {
            v16h vf = make_frag(sV + buf * 2048 + (nt * 16 + lm) * 32 + koff,
                                sV + buf * 2048 + (nt * 16 + lm) * 32 + 16 + koff);
            oacc[nt] = wmma16(pf, vf, oacc[nt]);
        }
    }
#pragma unroll
    for (int nt = 0; nt < 4; ++nt)
#pragma unroll
        for (int v = 0; v < 8; ++v) {
            int row = v + hf * 8;
            ctx[((size_t)(b * CS) + q0 + row) * CH + h * CHD + nt * 16 + lm] = oacc[nt][v];
        }
}

// ---------------- router: softmax over 8 experts, top-2 renormalized ----------------
__global__ __launch_bounds__(32) void router_kernel(const float* __restrict__ yn,
                                                    const float* __restrict__ Wr,
                                                    float* __restrict__ cw) {
    int t = blockIdx.x, lane = threadIdx.x;
    float acc[CE] = {};
    const float* row = yn + (size_t)t * CH;
    for (int i = lane; i < CH; i += 32) {
        float x = row[i];
        const float* wr = Wr + (size_t)i * CE;
#pragma unroll
        for (int e = 0; e < CE; ++e) acc[e] += x * wr[e];
    }
#pragma unroll
    for (int e = 0; e < CE; ++e)
#pragma unroll
        for (int off = 16; off >= 1; off >>= 1) acc[e] += __shfl_xor(acc[e], off, 32);

    if (lane == 0) {
        float mx = acc[0];
#pragma unroll
        for (int e = 1; e < CE; ++e) mx = fmaxf(mx, acc[e]);
        float p[CE], sum = 0.f;
#pragma unroll
        for (int e = 0; e < CE; ++e) { p[e] = __expf(acc[e] - mx); sum += p[e]; }
#pragma unroll
        for (int e = 0; e < CE; ++e) p[e] /= sum;
        int i1 = 0;
#pragma unroll
        for (int e = 1; e < CE; ++e) if (p[e] > p[i1]) i1 = e;
        int i2 = (i1 == 0) ? 1 : 0;
#pragma unroll
        for (int e = 0; e < CE; ++e) if (e != i1 && p[e] > p[i2]) i2 = e;
        float norm = p[i1] + p[i2];
#pragma unroll
        for (int e = 0; e < CE; ++e)
            cw[(size_t)t * CE + e] = (e == i1) ? p[i1] / norm : (e == i2) ? p[i2] / norm : 0.f;
    }
}

// ---------------- h = silu(g) * u ----------------
__global__ __launch_bounds__(256) void silu_mul_kernel(const float* __restrict__ g,
                                                       const float* __restrict__ u,
                                                       float* __restrict__ h, int n) {
    int i = blockIdx.x * 256 + threadIdx.x;
    if (i < n) {
        float gv = g[i];
        h[i] = gv / (1.f + __expf(-gv)) * u[i];
    }
}

// ---------------- launch ----------------
extern "C" void kernel_launch(void* const* d_in, const int* in_sizes, int n_in,
                              void* d_out, int out_size, void* d_ws, size_t ws_size,
                              hipStream_t stream) {
    const float* hidden = (const float*)d_in[0];
    const float* w_attn = (const float*)d_in[1];
    const float* Wq     = (const float*)d_in[2];
    const float* Wk     = (const float*)d_in[3];
    const float* Wv     = (const float*)d_in[4];
    const float* Wo     = (const float*)d_in[5];
    const float* w_moe  = (const float*)d_in[6];
    const float* Wr     = (const float*)d_in[7];
    const float* Wgate  = (const float*)d_in[8];
    const float* Wup    = (const float*)d_in[9];
    const float* Wdown  = (const float*)d_in[10];

    char* ws = (char*)d_ws;
    size_t off = 0;
    auto alloc = [&](size_t bytes) -> void* {
        void* p = ws + off;
        off += (bytes + 255) & ~(size_t)255;
        return p;
    };
    const size_t TH4 = (size_t)CT * CH * sizeof(float);
    const size_t TH2 = (size_t)CT * CH * sizeof(_Float16);
    const size_t TM4 = (size_t)CT * CM * sizeof(float);

    float*    xn   = (float*)alloc(TH4);
    float*    qbuf = (float*)alloc(TH4);
    float*    kbuf = (float*)alloc(TH4);
    float*    vbuf = (float*)alloc(TH4);
    _Float16* qr   = (_Float16*)alloc(TH2);
    _Float16* kr   = (_Float16*)alloc(TH2);
    _Float16* vt   = (_Float16*)alloc(TH2);
    float*    ctx  = (float*)alloc(TH4);
    float*    x1   = (float*)alloc(TH4);
    float*    yn   = (float*)alloc(TH4);
    float*    cw   = (float*)alloc((size_t)CT * CE * sizeof(float));
    float*    gbuf = (float*)alloc(TM4);
    float*    ubuf = (float*)alloc(TM4);
    float*    hbuf = (float*)alloc(TM4);
    (void)ws_size; (void)in_sizes; (void)n_in; (void)out_size;

    // 1. attention RMSNorm
    rmsnorm_kernel<<<CT, 256, 0, stream>>>(hidden, w_attn, xn, CH);

    // 2. Q/K/V projections (f16 WMMA GEMMs), 128x128 tiles
    dim3 gHH(CH / 128, CT / 128);
    gemm_kernel<0><<<gHH, 256, 0, stream>>>(xn, Wq, qbuf, CT, CH, CH, nullptr, nullptr, 0);
    gemm_kernel<0><<<gHH, 256, 0, stream>>>(xn, Wk, kbuf, CT, CH, CH, nullptr, nullptr, 0);
    gemm_kernel<0><<<gHH, 256, 0, stream>>>(xn, Wv, vbuf, CT, CH, CH, nullptr, nullptr, 0);

    // 3. RoPE + head layout + V transpose (f32 -> f16)
    rope_kernel<<<dim3(CS, CNH, CB), 64, 0, stream>>>(qbuf, kbuf, vbuf, qr, kr, vt);

    // 4. causal WMMA attention with async-LDS double buffering
    attn_kernel<<<dim3(CS / 16, CNH, CB), 32, ATTN_LDS, stream>>>(qr, kr, vt, ctx);

    // 5. O projection fused with residual add -> x1
    gemm_kernel<1><<<gHH, 256, 0, stream>>>(ctx, Wo, x1, CT, CH, CH, hidden, nullptr, 0);

    // 6. MoE RMSNorm + router
    rmsnorm_kernel<<<CT, 256, 0, stream>>>(x1, w_moe, yn, CH);
    router_kernel<<<CT, 32, 0, stream>>>(yn, Wr, cw);

    // 7. seed output with residual, then scale-accumulate each expert into it
    (void)hipMemcpyAsync(d_out, x1, TH4, hipMemcpyDeviceToDevice, stream);

    dim3 gHM(CM / 128, CT / 128);   // [T,H]x[H,M]
    dim3 gMH(CH / 128, CT / 128);   // [T,M]x[M,H]
    int nTM = CT * CM;
    for (int e = 0; e < CE; ++e) {
        gemm_kernel<0><<<gHM, 256, 0, stream>>>(yn, Wgate + (size_t)e * CH * CM, gbuf,
                                                CT, CM, CH, nullptr, nullptr, 0);
        gemm_kernel<0><<<gHM, 256, 0, stream>>>(yn, Wup + (size_t)e * CH * CM, ubuf,
                                                CT, CM, CH, nullptr, nullptr, 0);
        silu_mul_kernel<<<(nTM + 255) / 256, 256, 0, stream>>>(gbuf, ubuf, hbuf, nTM);
        gemm_kernel<2><<<gMH, 256, 0, stream>>>(hbuf, Wdown + (size_t)e * CM * CH, (float*)d_out,
                                                CT, CH, CM, nullptr, cw + e, CE);
    }
}
